// SurfacePrediction_57200374448507
// MI455X (gfx1250) — matrix operand order repaired and checked by reading, not compiled, not run on hardware
//
#include <hip/hip_runtime.h>
#include <stdint.h>

// ---------------------------------------------------------------------------
// CDNA5 (gfx1250) sparse U-Net forward.  Heavy work = gather-GEMM sparse convs
// mapped onto v_wmma_f32_16x16x32_bf16 (one WMMA K-step per 27/8-tap, f32 acc).
// B-fragments staged in LDS per workgroup; A rows prefetched one tap ahead.
// ---------------------------------------------------------------------------

typedef __attribute__((ext_vector_type(16))) __bf16       v16bf;
typedef __attribute__((ext_vector_type(8)))  float        v8f;
typedef __attribute__((ext_vector_type(4)))  unsigned int v4u;

#define BN_EPS 1e-4f

__device__ __forceinline__ unsigned short f2bf(float f) {
  unsigned int u = __float_as_uint(f);
  u += 0x7FFFu + ((u >> 16) & 1u);          // round-to-nearest-even
  return (unsigned short)(u >> 16);
}

// ---------------- p1: 27-tap conv Cin=3 -> 32 (tiny, fp32 VALU) -------------
__global__ __launch_bounds__(256)
void k_conv_p1(const float* __restrict__ feats, const float* __restrict__ W,
               const int* __restrict__ nbr, float* __restrict__ out, int N) {
  int t = blockIdx.x * blockDim.x + threadIdx.x;
  int m = t >> 5, d = t & 31;
  if (m >= N) return;
  float s = 0.f;
  for (int k = 0; k < 27; ++k) {
    int r = nbr[m * 27 + k];
    if (r < N) {                                  // r==N means inactive neighbor
      const float* f = feats + r * 3;
      const float* w = W + k * 96 + d;            // W[k][c][d], c-stride 32
      s += f[0] * w[0] + f[1] * w[32] + f[2] * w[64];
    }
  }
  out[m * 32 + d] = s;
}

// ---------------- BN statistics: scale/shift per channel --------------------
// Channel c comes from xa (c<Ca) or from gathered xg[parent[r]] (c>=Ca).
__global__ __launch_bounds__(256)
void k_bnstats(const float* __restrict__ xa, int Ca,
               const float* __restrict__ xg, int Cb, const int* __restrict__ parent,
               const float* __restrict__ g, const float* __restrict__ b,
               int N, float* __restrict__ scale, float* __restrict__ shift) {
  int c = blockIdx.x;
  __shared__ float ss[256], sq[256];
  float s = 0.f, q = 0.f;
  if (c < Ca) {
    for (int r = threadIdx.x; r < N; r += 256) { float v = xa[r * Ca + c]; s += v; q += v * v; }
  } else {
    int cc = c - Ca;
    for (int r = threadIdx.x; r < N; r += 256) { float v = xg[parent[r] * Cb + cc]; s += v; q += v * v; }
  }
  ss[threadIdx.x] = s; sq[threadIdx.x] = q;
  __syncthreads();
  for (int o = 128; o > 0; o >>= 1) {
    if (threadIdx.x < o) { ss[threadIdx.x] += ss[threadIdx.x + o]; sq[threadIdx.x] += sq[threadIdx.x + o]; }
    __syncthreads();
  }
  if (threadIdx.x == 0) {
    float inv = 1.f / (float)N;
    float mean = ss[0] * inv;
    float var  = sq[0] * inv - mean * mean;       // biased var (jnp.var)
    float sc   = g[c] / sqrtf(var + BN_EPS);
    scale[c] = sc;
    shift[c] = b[c] - mean * sc;
  }
}

// ---------------- fused BN+ReLU -> bf16 activations (+zero pad row) ---------
__global__ __launch_bounds__(256)
void k_bnrelu_bf16(const float* __restrict__ x, const float* __restrict__ scale,
                   const float* __restrict__ shift, unsigned short* __restrict__ yb, int N) {
  int i = blockIdx.x * blockDim.x + threadIdx.x;
  if (i >= (N + 1) * 32) return;
  int r = i >> 5, c = i & 31;
  float v = 0.f;
  if (r < N) { v = x[i] * scale[c] + shift[c]; v = v > 0.f ? v : 0.f; }
  yb[i] = f2bf(v);
}

// ---------------- in-place BN+ReLU (fp32, decoder path) ---------------------
__global__ __launch_bounds__(256)
void k_bnrelu_f32(float* __restrict__ x, const float* __restrict__ scale,
                  const float* __restrict__ shift, int total, int C) {
  int i = blockIdx.x * blockDim.x + threadIdx.x;
  if (i >= total) return;
  int c = i % C;
  float v = x[i] * scale[c] + shift[c];
  x[i] = v > 0.f ? v : 0.f;
}

// ---------------- weight -> WMMA B-fragment swizzle (bf16) ------------------
// bfrag[((k*2+t)*32 + lane)*16 + j] = bf16(W[k][ j + 16*(lane>>4) ][ 16*t + (lane&15) ])
// (B 32x16, contiguous k order used by this code, not AMD's table order: lane = column N (+16 for K-high half), VGPR v holds K=2v,2v+1)
__global__ __launch_bounds__(256)
void k_prep_bfrag(const float* __restrict__ W, unsigned short* __restrict__ bf, int taps) {
  int i = blockIdx.x * blockDim.x + threadIdx.x;
  if (i >= taps * 1024) return;
  int j  = i & 15;
  int l  = (i >> 4) & 31;
  int kt = i >> 9;
  int k  = kt >> 1, t = kt & 1;
  int K  = j + 16 * (l >> 4);
  int Nc = 16 * t + (l & 15);
  bf[i] = f2bf(W[(k * 32 + K) * 32 + Nc]);
}

// ---------------- gather-GEMM sparse conv via WMMA --------------------------
// One wave = 16-row x 32-col output tile.  Per tap: one bf16 A-frag gather
// (2 x b128 per lane, prefetched one tap ahead) + LDS-resident B-frags
// -> two v_wmma_f32_16x16x32_bf16.  Fully unrolled (TAPS is 27 or 8).
union FragU { v4u u[2]; v16bf v; };
union AccU  { v8f v; float f[8]; };

template <int TAPS, bool RESID>
__global__ __launch_bounds__(256)
void k_gconv_wmma(const unsigned short* __restrict__ xb,   // [(NP+1)*32] bf16, row NP = zeros
                  const unsigned short* __restrict__ bfrag,// [TAPS*2*32*16] bf16 B-fragments
                  const int* __restrict__ idx,             // [M*TAPS], value NP = missing
                  const float* __restrict__ resid,         // residual (may alias out)
                  float* __restrict__ out,                 // [M*32]
                  int M, int NP) {
  extern __shared__ unsigned short ldsB[];                 // TAPS*1024 bf16

  // cooperative copy of all B-fragments into LDS (shared by the 8 waves)
  {
    const v4u* src = (const v4u*)bfrag;
    v4u*       dst = (v4u*)ldsB;
    for (int i = threadIdx.x; i < TAPS * 128; i += 256) dst[i] = src[i];
  }
  __syncthreads();

  int wave = (blockIdx.x * blockDim.x + threadIdx.x) >> 5;
  int row0 = wave * 16;
  if (row0 >= M) return;                                   // wave-uniform
  int lane = threadIdx.x & 31;
  int half = lane >> 4;                                    // K-half for A, M-half for C
  int lr   = lane & 15;                                    // A row / C column within tile

  int  arow   = row0 + lr;
  bool avalid = arow < M;
  int  asafe  = avalid ? arow : 0;
  const int* myidx = idx + (size_t)asafe * TAPS;

  AccU a0, a1;
  if (RESID) {
    for (int r = 0; r < 8; ++r) {
      int m = row0 + r + 8 * half;
      a0.f[r] = (m < M) ? resid[m * 32 + lr]      : 0.f;
      a1.f[r] = (m < M) ? resid[m * 32 + 16 + lr] : 0.f;
    }
  } else {
    for (int r = 0; r < 8; ++r) { a0.f[r] = 0.f; a1.f[r] = 0.f; }
  }

  int rr = avalid ? myidx[0] : NP;                         // row for tap 0
  __builtin_prefetch(xb + (size_t)rr * 32, 0, 3);          // global_prefetch

#pragma unroll
  for (int k = 0; k < TAPS; ++k) {
    int rcur = rr;
    if (k + 1 < TAPS) {                                    // pipeline next tap's gather
      int t = myidx[k + 1];
      rr = avalid ? t : NP;
      __builtin_prefetch(xb + (size_t)rr * 32, 0, 3);
    }
    const v4u* pa = (const v4u*)(xb + (size_t)rcur * 32);
    FragU A;
    A.u[0] = pa[2 * half];                                 // K = 16h .. 16h+7
    A.u[1] = pa[2 * half + 1];                             // K = 16h+8 .. 16h+15
    const v4u* pb = (const v4u*)(ldsB + (((size_t)k * 2) * 32 + lane) * 16);
    FragU B0, B1;
    B0.u[0] = pb[0];  B0.u[1] = pb[1];                     // cols 0..15
    B1.u[0] = pb[64]; B1.u[1] = pb[65];                    // cols 16..31 (next 32x16 tile)
    a0.v = __builtin_amdgcn_wmma_f32_16x16x32_bf16(false, A.v, false, B0.v, (short)0, a0.v, false, false);
    a1.v = __builtin_amdgcn_wmma_f32_16x16x32_bf16(false, A.v, false, B1.v, (short)0, a1.v, false, false);
  }

  for (int r = 0; r < 8; ++r) {
    int m = row0 + r + 8 * half;
    if (m < M) {
      out[m * 32 + lr]      = a0.f[r];
      out[m * 32 + 16 + lr] = a1.f[r];
    }
  }
}

// ---------------- unpool + concat (decoder) ---------------------------------
__global__ __launch_bounds__(256)
void k_concat(const float* __restrict__ xa, const float* __restrict__ xg,
              const int* __restrict__ parent, float* __restrict__ out,
              int N, int Ca, int Cb) {
  int i = blockIdx.x * blockDim.x + threadIdx.x;
  int C = Ca + Cb;
  if (i >= N * C) return;
  int r = i / C, c = i % C;
  out[i] = (c < Ca) ? xa[r * Ca + c] : xg[parent[r] * Cb + (c - Ca)];
}

// ---------------- fused p3 (BN96+ReLU on virtual concat) + linear 96->1 -----
__global__ __launch_bounds__(256)
void k_final(const float* __restrict__ x0, const float* __restrict__ y1,
             const int* __restrict__ parent0, const float* __restrict__ scale,
             const float* __restrict__ shift, const float* __restrict__ Wlin,
             const float* __restrict__ blin, float* __restrict__ out, int N0) {
  int r = blockIdx.x * blockDim.x + threadIdx.x;
  if (r >= N0) return;
  float s = blin[0];
  int p = parent0[r];
  for (int c = 0; c < 32; ++c) {
    float v = x0[r * 32 + c] * scale[c] + shift[c];
    s += (v > 0.f ? v : 0.f) * Wlin[c];
  }
  for (int c = 0; c < 64; ++c) {
    float v = y1[p * 64 + c] * scale[32 + c] + shift[32 + c];
    s += (v > 0.f ? v : 0.f) * Wlin[32 + c];
  }
  out[r] = s;
}

// ---------------------------------------------------------------------------
extern "C" void kernel_launch(void* const* d_in, const int* in_sizes, int n_in,
                              void* d_out, int out_size, void* d_ws, size_t ws_size,
                              hipStream_t stream) {
  (void)n_in; (void)ws_size; (void)out_size;
  // inputs in setup_inputs() dict order; params in dict-insertion order
  const float* feats = (const float*)d_in[0];
  const float* W_p1  = (const float*)d_in[1];
  struct Lvl { const float *g1, *b1, *W1, *g2, *b2, *W2; } L[3];
  for (int i = 0; i < 3; ++i) {
    int k = 2 + i * 6;
    L[i].g1 = (const float*)d_in[k + 0]; L[i].b1 = (const float*)d_in[k + 1];
    L[i].W1 = (const float*)d_in[k + 2]; L[i].g2 = (const float*)d_in[k + 3];
    L[i].b2 = (const float*)d_in[k + 4]; L[i].W2 = (const float*)d_in[k + 5];
  }
  const float* gd0 = (const float*)d_in[20]; const float* bd0 = (const float*)d_in[21];
  const float* Wdown0 = (const float*)d_in[22];
  const float* gd1 = (const float*)d_in[23]; const float* bd1 = (const float*)d_in[24];
  const float* Wdown1 = (const float*)d_in[25];
  const float* gu2 = (const float*)d_in[26]; const float* bu2 = (const float*)d_in[27];
  const float* gu1 = (const float*)d_in[28]; const float* bu1 = (const float*)d_in[29];
  const float* g3  = (const float*)d_in[30]; const float* b3  = (const float*)d_in[31];
  const float* Wlin = (const float*)d_in[32]; const float* blin = (const float*)d_in[33];
  const int* nbr0 = (const int*)d_in[34];
  const int* nbr1 = (const int*)d_in[35];
  const int* nbr2 = (const int*)d_in[36];
  const int* child0 = (const int*)d_in[37];
  const int* child1 = (const int*)d_in[38];
  const int* parent0 = (const int*)d_in[39];
  const int* parent1 = (const int*)d_in[40];

  const int N0 = in_sizes[0] / 3;
  const int N1 = in_sizes[37] / 8;
  const int N2 = in_sizes[38] / 8;

  // bump allocator over workspace
  char* w = (char*)d_ws;
  auto alloc = [&](size_t bytes) -> void* {
    void* p = (void*)w;
    w += (bytes + 255) & ~(size_t)255;
    return p;
  };
  float* x0  = (float*)alloc((size_t)N0 * 32 * 4);
  float* h   = (float*)alloc((size_t)N0 * 32 * 4);          // res-block temp (max level-0 size)
  float* x1  = (float*)alloc((size_t)N1 * 32 * 4);
  float* x2  = (float*)alloc((size_t)N2 * 32 * 4);
  float* y1b = (float*)alloc((size_t)N1 * 64 * 4);          // concat(x1, y2[parent1]) -> y1
  unsigned short* xb    = (unsigned short*)alloc(((size_t)N0 + 1) * 32 * 2); // bf16 acts + pad row
  unsigned short* bfrag = (unsigned short*)alloc((size_t)27 * 1024 * 2);     // B fragments (max taps)
  float* scl = (float*)alloc(96 * 4);
  float* shf = (float*)alloc(96 * 4);

  auto cdiv = [](int a, int b) { return (a + b - 1) / b; };
  auto stats = [&](const float* x, int C, const float* g, const float* b, int N) {
    k_bnstats<<<C, 256, 0, stream>>>(x, C, nullptr, 0, nullptr, g, b, N, scl, shf);
  };
  auto tobf = [&](const float* x, int N) {
    k_bnrelu_bf16<<<cdiv((N + 1) * 32, 256), 256, 0, stream>>>(x, scl, shf, xb, N);
  };
  auto gconv = [&](const float* W, int taps, const int* idx, const float* resid,
                   float* outp, int M, int NP) {
    k_prep_bfrag<<<cdiv(taps * 1024, 256), 256, 0, stream>>>(W, bfrag, taps);
    int blocks = cdiv(cdiv(M, 16), 8);               // 8 waves/block, 1 tile/wave
    size_t shmem = (size_t)taps * 2048;              // B-fragments in LDS
    if (taps == 27) {
      if (resid)
        k_gconv_wmma<27, true><<<blocks, 256, shmem, stream>>>(xb, bfrag, idx, resid, outp, M, NP);
      else
        k_gconv_wmma<27, false><<<blocks, 256, shmem, stream>>>(xb, bfrag, idx, nullptr, outp, M, NP);
    } else {
      k_gconv_wmma<8, false><<<blocks, 256, shmem, stream>>>(xb, bfrag, idx, nullptr, outp, M, NP);
    }
  };
  auto resblock = [&](float* x, int N, const int* nbr, const Lvl& P) {
    stats(x, 32, P.g1, P.b1, N); tobf(x, N);
    gconv(P.W1, 27, nbr, nullptr, h, N, N);
    stats(h, 32, P.g2, P.b2, N); tobf(h, N);
    gconv(P.W2, 27, nbr, x, x, N, N);        // fused identity residual (safe self-alias)
  };

  // ---- encoder ----
  k_conv_p1<<<cdiv(N0 * 32, 256), 256, 0, stream>>>(feats, W_p1, nbr0, x0, N0);
  resblock(x0, N0, nbr0, L[0]);
  stats(x0, 32, gd0, bd0, N0); tobf(x0, N0);
  gconv(Wdown0, 8, child0, nullptr, x1, N1, N0);
  resblock(x1, N1, nbr1, L[1]);
  stats(x1, 32, gd1, bd1, N1); tobf(x1, N1);
  gconv(Wdown1, 8, child1, nullptr, x2, N2, N1);
  resblock(x2, N2, nbr2, L[2]);

  // ---- decoder ----
  stats(x2, 32, gu2, bu2, N2);
  k_bnrelu_f32<<<cdiv(N2 * 32, 256), 256, 0, stream>>>(x2, scl, shf, N2 * 32, 32);   // x2 := y2
  k_concat<<<cdiv(N1 * 64, 256), 256, 0, stream>>>(x1, x2, parent1, y1b, N1, 32, 32);
  k_bnstats<<<64, 256, 0, stream>>>(y1b, 64, nullptr, 0, nullptr, gu1, bu1, N1, scl, shf);
  k_bnrelu_f32<<<cdiv(N1 * 64, 256), 256, 0, stream>>>(y1b, scl, shf, N1 * 64, 64);  // y1b := y1
  // BN96 stats over virtual concat(x0, y1[parent0]) — never materialized
  k_bnstats<<<96, 256, 0, stream>>>(x0, 32, y1b, 64, parent0, g3, b3, N0, scl, shf);
  k_final<<<cdiv(N0, 256), 256, 0, stream>>>(x0, y1b, parent0, scl, shf, Wlin, blin,
                                             (float*)d_out, N0);
}
